// BilinearPairedLayer_5162550689975
// MI455X (gfx1250) — matrix operand order repaired and checked by reading, not compiled, not run on hardware
//
#include <hip/hip_runtime.h>
#include <hip/hip_bf16.h>

// ---------------------------------------------------------------------------
// BilinearPairedLayer for MI455X (gfx1250, wave32, WMMA)
//
// Dominant cost: 16 GEMMs of 512x512x128 (1.07 GFLOP) -> v_wmma_f32_16x16x32_bf16
// with an hi/lo bf16 error-compensated split (Ah*Bh + Ah*Bl + Al*Bh) for
// near-fp32 accuracy at 16-bit matrix-pipe rates.
// ---------------------------------------------------------------------------

typedef __attribute__((ext_vector_type(16))) __bf16 v16bf;
typedef __attribute__((ext_vector_type(8)))  float  v8f;

union FragU { v16bf v; uint4 q[2]; };

#define BB  2
#define NN  512
#define NI  64
#define FF  192   // NI * context
#define HH  128
#define OO  8

// ---- bf16 helpers (round-to-nearest-even) ----------------------------------
__device__ __forceinline__ unsigned short f2bf(float x) {
    union { float f; unsigned u; } v; v.f = x;
    unsigned u = v.u;
    unsigned r = u + 0x7FFFu + ((u >> 16) & 1u);
    return (unsigned short)(r >> 16);
}
__device__ __forceinline__ float bf2f(unsigned short h) {
    union { unsigned u; float f; } v; v.u = ((unsigned)h) << 16;
    return v.f;
}
__device__ __forceinline__ void split_bf(float x, unsigned short* hi, unsigned short* lo) {
    unsigned short h = f2bf(x);
    *hi = h;
    *lo = f2bf(x - bf2f(h));
}

// ---------------------------------------------------------------------------
// Kernel 1: context pairing + fc + CELU.
//   zl = [x_l[n], x_l[n-1], x_l[n+1]]
//   zr = [x_l[n], x_r[n+1], x_r[n-1]]   (z_r seeded with x_l: bug-faithful)
// Stores hl fp32 (consumed in fp32 by kernel 2) and hr as bf16 hi/lo.
// One block per (b,n); 128 threads = one output channel h each.
// ---------------------------------------------------------------------------
__global__ __launch_bounds__(HH) void pairing_fc_celu(
    const float* __restrict__ xl, const float* __restrict__ xr,
    const float* __restrict__ Wl, const float* __restrict__ bl,
    const float* __restrict__ Wr, const float* __restrict__ br,
    float* __restrict__ hl,
    unsigned short* __restrict__ hr_hi, unsigned short* __restrict__ hr_lo)
{
    __shared__ float zl[FF];
    __shared__ float zr[FF];

    const int bn = blockIdx.x;            // bn = b*NN + n
    const int b  = bn >> 9;
    const int n  = bn & (NN - 1);
    const float* xlb = xl + (size_t)b * NN * NI;
    const float* xrb = xr + (size_t)b * NN * NI;

    for (int f = threadIdx.x; f < FF; f += blockDim.x) {
        const int seg = f >> 6;           // 0,1,2
        const int c   = f & (NI - 1);
        float vl, vr;
        if (seg == 0) {                   // both lists seeded with x_l
            vl = xlb[n * NI + c];
            vr = vl;
        } else if (seg == 1) {            // zl: shift_fwd(x_l)  -> x_l[n-1]
                                          // zr: shift_bwd(x_r)  -> x_r[n+1]
            vl = (n > 0)      ? xlb[(n - 1) * NI + c] : 0.f;
            vr = (n < NN - 1) ? xrb[(n + 1) * NI + c] : 0.f;
        } else {                          // zl: shift_bwd(x_l)  -> x_l[n+1]
                                          // zr: shift_fwd(x_r)  -> x_r[n-1]
            vl = (n < NN - 1) ? xlb[(n + 1) * NI + c] : 0.f;
            vr = (n > 0)      ? xrb[(n - 1) * NI + c] : 0.f;
        }
        zl[f] = vl;
        zr[f] = vr;
    }
    __syncthreads();

    const int h = threadIdx.x;
    float al = bl[h];
    float ar = br[h];
    const float* wlr = Wl + (size_t)h * FF;
    const float* wrr = Wr + (size_t)h * FF;
    #pragma unroll 4
    for (int f = 0; f < FF; ++f) {
        al = fmaf(zl[f], wlr[f], al);
        ar = fmaf(zr[f], wrr[f], ar);
    }
    // CELU(alpha=1): x>0 ? x : expm1(x)
    al = (al > 0.f) ? al : expm1f(al);
    ar = (ar > 0.f) ? ar : expm1f(ar);

    const size_t idx = (size_t)bn * HH + h;
    hl[idx] = al;
    split_bf(ar, &hr_hi[idx], &hr_lo[idx]);
}

// ---------------------------------------------------------------------------
// Kernel 2: t[b,o,i,g] = sum_h hl[b,i,h] * W[o,h,g], fp32 math, bf16 hi/lo out.
// One block per (b,o,i); 128 threads = one g each; hl row staged in LDS.
// ---------------------------------------------------------------------------
__global__ __launch_bounds__(HH) void make_t(
    const float* __restrict__ hl, const float* __restrict__ W,
    unsigned short* __restrict__ t_hi, unsigned short* __restrict__ t_lo)
{
    __shared__ float sh[HH];
    const int boi = blockIdx.x;           // boi = (b*OO + o)*NN + i
    const int i   = boi & (NN - 1);
    const int bo  = boi >> 9;
    const int o   = bo & (OO - 1);
    const int b   = bo >> 3;

    const int g = threadIdx.x;
    sh[g] = hl[((size_t)(b * NN + i)) * HH + g];
    __syncthreads();

    const float* Wo = W + (size_t)o * HH * HH;
    float acc = 0.f;
    #pragma unroll 4
    for (int h = 0; h < HH; ++h)
        acc = fmaf(sh[h], Wo[(size_t)h * HH + g], acc);

    const size_t idx = (size_t)boi * HH + g;   // layout [b][o][i][g]
    split_bf(acc, &t_hi[idx], &t_lo[idx]);
}

// ---------------------------------------------------------------------------
// Kernel 3: out[b,i,j,o] = sum_g t[b,o,i,g] * hr[b,j,g] + bias[o]
// 16 GEMMs (M=N=512, K=128). Block = 128 threads (4 waves) -> 64x64 tile of
// one (b,o) GEMM; wave w owns rows [w*16,w*16+16) x 64 cols = 4 C fragments.
// K split: acc += Ah*Bh + Ah*Bl + Al*Bh per 16x16x32 step.
//
// Fragment addressing (A and B share a contiguous k order, not AMD's table order):
//   A(16x32): lane L row = L%16; K = k0 + (L>=16)*16 .. +15
//             -> two 16B loads.
//   B(32x16): lane L col = L%16; K = k0 + (L>=16)*16 .. +15 -> two 16B loads.
//   C(16x16): vgpr r, lane L -> M = r + (L>=16)*8, N = L%16.
// ---------------------------------------------------------------------------
__global__ __launch_bounds__(128) void bilinear_gemm_wmma(
    const unsigned short* __restrict__ t_hi, const unsigned short* __restrict__ t_lo,
    const unsigned short* __restrict__ hr_hi, const unsigned short* __restrict__ hr_lo,
    const float* __restrict__ bias, float* __restrict__ out)
{
    const int lane = threadIdx.x & 31;
    const int wv   = threadIdx.x >> 5;
    const int bo   = blockIdx.z;          // b*OO + o
    const int b    = bo >> 3;
    const int o    = bo & (OO - 1);
    const int lm   = lane & 15;
    const int lh   = lane >> 4;           // half-wave select
    const int m0   = blockIdx.y * 64 + wv * 16;
    const int n0   = blockIdx.x * 64;

    const unsigned short* aH = t_hi + ((size_t)bo * NN + m0 + lm) * HH;
    const unsigned short* aL = t_lo + ((size_t)bo * NN + m0 + lm) * HH;
    const unsigned short* bH = hr_hi + ((size_t)b * NN + n0 + lm) * HH;
    const unsigned short* bL = hr_lo + ((size_t)b * NN + n0 + lm) * HH;

    v8f acc[4];
    #pragma unroll
    for (int c = 0; c < 4; ++c)
        acc[c] = (v8f){0.f, 0.f, 0.f, 0.f, 0.f, 0.f, 0.f, 0.f};

    #pragma unroll
    for (int ks = 0; ks < 4; ++ks) {
        const int k0 = ks * 32;
        FragU Ah, Al;
        const int ak = k0 + lh * 16;
        Ah.q[0] = *(const uint4*)(aH + ak);
        Ah.q[1] = *(const uint4*)(aH + ak + 8);
        Al.q[0] = *(const uint4*)(aL + ak);
        Al.q[1] = *(const uint4*)(aL + ak + 8);

        const int bk = k0 + lh * 16;
        #pragma unroll
        for (int c = 0; c < 4; ++c) {
            FragU Bh, Bl;
            const unsigned short* pH = bH + (size_t)c * 16 * HH + bk;
            const unsigned short* pL = bL + (size_t)c * 16 * HH + bk;
            Bh.q[0] = *(const uint4*)pH;
            Bh.q[1] = *(const uint4*)(pH + 8);
            Bl.q[0] = *(const uint4*)pL;
            Bl.q[1] = *(const uint4*)(pL + 8);

            acc[c] = __builtin_amdgcn_wmma_f32_16x16x32_bf16(
                false, Ah.v, false, Bh.v, (short)0, acc[c], false, false);
            acc[c] = __builtin_amdgcn_wmma_f32_16x16x32_bf16(
                false, Ah.v, false, Bl.v, (short)0, acc[c], false, false);
            acc[c] = __builtin_amdgcn_wmma_f32_16x16x32_bf16(
                false, Al.v, false, Bh.v, (short)0, acc[c], false, false);
        }
    }

    const float bb = bias[o];
    #pragma unroll
    for (int c = 0; c < 4; ++c) {
        const int j = n0 + c * 16 + lm;
        #pragma unroll
        for (int r = 0; r < 8; ++r) {
            const int i = m0 + r + lh * 8;
            out[(((size_t)b * NN + i) * NN + j) * OO + o] = acc[c][r] + bb;
        }
    }
}

// ---------------------------------------------------------------------------
extern "C" void kernel_launch(void* const* d_in, const int* in_sizes, int n_in,
                              void* d_out, int out_size, void* d_ws, size_t ws_size,
                              hipStream_t stream) {
    (void)in_sizes; (void)n_in; (void)out_size; (void)ws_size;

    const float* x_l  = (const float*)d_in[0];
    const float* x_r  = (const float*)d_in[1];
    const float* Wl   = (const float*)d_in[2];
    const float* bl   = (const float*)d_in[3];
    const float* Wr   = (const float*)d_in[4];
    const float* br   = (const float*)d_in[5];
    const float* bw   = (const float*)d_in[6];   // [OO,HH,HH]
    const float* bb   = (const float*)d_in[7];   // [OO]
    float* out        = (float*)d_out;

    // Workspace carving (~5 MB, 16B-aligned chunks)
    char* ws = (char*)d_ws;
    float*          hl    = (float*)ws;                      // BB*NN*HH f32  = 512 KB
    ws += (size_t)BB * NN * HH * sizeof(float);
    unsigned short* hr_hi = (unsigned short*)ws;             // BB*NN*HH bf16 = 256 KB
    ws += (size_t)BB * NN * HH * sizeof(unsigned short);
    unsigned short* hr_lo = (unsigned short*)ws;
    ws += (size_t)BB * NN * HH * sizeof(unsigned short);
    unsigned short* t_hi  = (unsigned short*)ws;             // BB*OO*NN*HH bf16 = 2 MB
    ws += (size_t)BB * OO * NN * HH * sizeof(unsigned short);
    unsigned short* t_lo  = (unsigned short*)ws;

    // Phase 1: pairing + fc + CELU  (1024 blocks x 128 threads)
    pairing_fc_celu<<<BB * NN, HH, 0, stream>>>(
        x_l, x_r, Wl, bl, Wr, br, hl, hr_hi, hr_lo);

    // Phase 2: t = hl @ W[o]  (8192 blocks x 128 threads)
    make_t<<<BB * OO * NN, HH, 0, stream>>>(hl, bw, t_hi, t_lo);

    // Phase 3: pair-grid GEMM via WMMA bf16 hi/lo split
    dim3 grid(NN / 64, NN / 64, BB * OO);  // (8, 8, 16)
    bilinear_gemm_wmma<<<grid, 128, 0, stream>>>(
        t_hi, t_lo, hr_hi, hr_lo, bb, out);
}
